// GRACEGConv_26345329393832
// MI455X (gfx1250) — compile-verified, matrix operand order checked by reading, not run on hardware
//
#include <hip/hip_runtime.h>
#include <stdint.h>

typedef __attribute__((ext_vector_type(16))) _Float16 v16h;
typedef __attribute__((ext_vector_type(8)))  _Float16 v8h;
typedef __attribute__((ext_vector_type(8)))  float    v8f;

#define F 128           // feature dim (IN_DIM == HID_DIM == 128)
#define TPB 256
#define TILES_PER_BLOCK 8   // 8 waves/block, one 16-row tile per wave

// ---------------- utility kernels ----------------

__global__ void k_fill1(float* __restrict__ p, int n) {
    int i = blockIdx.x * blockDim.x + threadIdx.x;
    if (i < n) p[i] = 1.0f;   // self-loop contributes 1 to every degree
}

__global__ void k_zero(float* __restrict__ p, int n) {
    int i = blockIdx.x * blockDim.x + threadIdx.x;
    if (i < n) p[i] = 0.0f;
}

__global__ void k_degree(const int* __restrict__ dst, float* __restrict__ deg, int E) {
    int i = blockIdx.x * blockDim.x + threadIdx.x;
    if (i < E) atomicAdd(&deg[dst[i]], 1.0f);
}

__global__ void k_rsqrt(const float* __restrict__ deg, float* __restrict__ dinv, int n) {
    int i = blockIdx.x * blockDim.x + threadIdx.x;
    if (i < n) dinv[i] = rsqrtf(deg[i]);     // deg >= 1 always (self-loop)
}

__global__ void k_cvt_half(const float* __restrict__ in, _Float16* __restrict__ out, int n) {
    int i = blockIdx.x * blockDim.x + threadIdx.x;
    if (i < n) out[i] = (_Float16)in[i];
}

// W is [K][N] row-major fp32; produce WT as [N][K] fp16 so B fragments are
// contiguous 16B chunks in LDS (and stageable verbatim via async-to-LDS).
__global__ void k_cvt_WT(const float* __restrict__ W, _Float16* __restrict__ WT) {
    int i = blockIdx.x * blockDim.x + threadIdx.x;   // i = k*128 + n (coalesced read)
    if (i < F * F) {
        int k = i >> 7, n = i & 127;
        WT[n * F + k] = (_Float16)W[i];
    }
}

// ---------------- WMMA GEMM: C[M,128] = A[M,128] x W[128,128] ----------------
// A is fp16 row-major, WT is fp16 [N][K] in global (staged to LDS async).
__global__ void __launch_bounds__(TPB)
k_gemm_wmma(const _Float16* __restrict__ A, const _Float16* __restrict__ WT,
            float* __restrict__ C, int M) {
    __shared__ _Float16 sW[F * F];     // 32 KB, [N][K]

    const int tid = threadIdx.x;

    // Stage entire 32KB WT tile via gfx1250 async global->LDS copy (ASYNCcnt).
    {
        unsigned      lbase = (unsigned)(uintptr_t)&sW[0];     // low 32 bits == LDS offset
        unsigned long gbase = (unsigned long)(uintptr_t)WT;
#pragma unroll
        for (int r = 0; r < 8; ++r) {
            unsigned off = (unsigned)(r * 4096 + tid * 16);    // 256 thr * 16B * 8 = 32KB
            asm volatile("global_load_async_to_lds_b128 %0, %1, off"
                         :: "v"(lbase + off), "v"(gbase + (unsigned long)off)
                         : "memory");
        }
        asm volatile("s_wait_asynccnt 0" ::: "memory");
    }
    __syncthreads();

    const int wave = tid >> 5;                 // wave32
    const int lane = tid & 31;
    const int tile = blockIdx.x * TILES_PER_BLOCK + wave;   // 16-row tile index

    if (tile * 16 < M) {
        const int hi   = lane >> 4;            // 0: lanes 0-15, 1: lanes 16-31
        const int l16  = lane & 15;
        const int row  = tile * 16 + l16;      // A-matrix row for this lane
        const int koff = hi * 8;               // ISA 16-bit A layout K shift

        v8f zero = {};
        v8f acc[8];
#pragma unroll
        for (int nt = 0; nt < 8; ++nt) acc[nt] = zero;

#pragma unroll
        for (int kb = 0; kb < 4; ++kb) {       // K = 128 in 4 steps of 32
            const int kbase = kb * 32;
            // A fragment 16x32: lane holds K = kbase+koff+[0..7] and +[16..23]
            const _Float16* ap = A + (size_t)row * F + kbase + koff;
            v8h a0 = *(const v8h*)(ap);
            v8h a1 = *(const v8h*)(ap + 16);
            v16h a;
#pragma unroll
            for (int i = 0; i < 8; ++i) { a[i] = a0[i]; a[8 + i] = a1[i]; }

#pragma unroll
            for (int nt = 0; nt < 8; ++nt) {   // 8 column tiles -> full 128 cols
                // B fragment 32x16: lane holds column n, same K striping as A
                const _Float16* bp = &sW[(size_t)(nt * 16 + l16) * F + kbase + koff];
                v8h b0 = *(const v8h*)(bp);
                v8h b1 = *(const v8h*)(bp + 16);
                v16h b;
#pragma unroll
                for (int i = 0; i < 8; ++i) { b[i] = b0[i]; b[8 + i] = b1[i]; }

                acc[nt] = __builtin_amdgcn_wmma_f32_16x16x32_f16(
                    false, a, false, b, (short)0, acc[nt], false, false);
            }
        }

        // D layout: VGPR v -> (M = v + 8*hi, N = l16) within the 16x16 tile
#pragma unroll
        for (int nt = 0; nt < 8; ++nt) {
            float* cp = C + (size_t)(tile * 16 + 8 * hi) * F + nt * 16 + l16;
#pragma unroll
            for (int v = 0; v < 8; ++v) cp[(size_t)v * F] = acc[nt][v];
        }
    }
}

// ---------------- message scatter: agg[dst] += h[src] * dinv[s]*dinv[d] ----------------
// 32 lanes per edge, float4 per lane -> fully coalesced 512B gather per edge.
__global__ void k_edge_scatter(const int* __restrict__ src, const int* __restrict__ dst,
                               const float* __restrict__ dinv, const float* __restrict__ h,
                               float* __restrict__ agg, int E) {
    int gid = blockIdx.x * blockDim.x + threadIdx.x;
    int e = gid >> 5;
    int l = gid & 31;
    if (e < E) {
        int s = src[e], d = dst[e];
        float w = dinv[s] * dinv[d];
        const float4 hv = *(const float4*)(h + (size_t)s * F + l * 4);
        float* ap = agg + (size_t)d * F + l * 4;
        atomicAdd(ap + 0, hv.x * w);
        atomicAdd(ap + 1, hv.y * w);
        atomicAdd(ap + 2, hv.z * w);
        atomicAdd(ap + 3, hv.w * w);
    }
}

// self-loop: agg[i] += h[i] * dinv[i]^2  (runs after scatter, no atomics needed)
__global__ void k_self_loop(const float* __restrict__ dinv, const float* __restrict__ h,
                            float* __restrict__ agg, int n) {
    int i = blockIdx.x * blockDim.x + threadIdx.x;
    if (i < n) {
        float di = dinv[i >> 7];
        agg[i] += h[i] * di * di;
    }
}

// out = relu(agg + b); optionally also emit fp16 copy for next layer's A operand
__global__ void k_bias_relu(const float* __restrict__ agg, const float* __restrict__ b,
                            float* __restrict__ zf, _Float16* __restrict__ zh, int n) {
    int i = blockIdx.x * blockDim.x + threadIdx.x;
    if (i < n) {
        float v = agg[i] + b[i & 127];
        v = v > 0.0f ? v : 0.0f;
        zf[i] = v;
        if (zh) zh[i] = (_Float16)v;
    }
}

// ---------------- launcher ----------------

extern "C" void kernel_launch(void* const* d_in, const int* in_sizes, int n_in,
                              void* d_out, int out_size, void* d_ws, size_t ws_size,
                              hipStream_t stream) {
    const float* x  = (const float*)d_in[0];
    const int*   ei = (const int*)d_in[1];
    const float* W1 = (const float*)d_in[2];
    const float* b1 = (const float*)d_in[3];
    const float* W2 = (const float*)d_in[4];
    const float* b2 = (const float*)d_in[5];

    const int N  = in_sizes[0] / F;     // 50000
    const int E  = in_sizes[1] / 2;     // 600000
    const int NF = N * F;               // 6.4M

    const int* src = ei;
    const int* dst = ei + E;

    // workspace carve-up
    float*    deg  = (float*)d_ws;                 // N
    float*    dinv = deg + N;                      // N
    float*    h    = dinv + N;                     // N*F  (GEMM out, then layer-1 z)
    float*    agg  = h + NF;                       // N*F
    _Float16* Ah   = (_Float16*)(agg + NF);        // N*F fp16 A operand
    _Float16* WT   = Ah + NF;                      // F*F fp16 transposed weights

    const int blkN  = (N + TPB - 1) / TPB;
    const int blkNF = (NF + TPB - 1) / TPB;
    const int blkE  = (E + TPB - 1) / TPB;
    const int blkEdge = (E * 32 + TPB - 1) / TPB;
    const int tiles = (N + 15) / 16;
    const int blkGemm = (tiles + TILES_PER_BLOCK - 1) / TILES_PER_BLOCK;
    const int blkW  = (F * F + TPB - 1) / TPB;

    // --- normalization (shared by both layers) ---
    k_fill1<<<blkN, TPB, 0, stream>>>(deg, N);
    k_degree<<<blkE, TPB, 0, stream>>>(dst, deg, E);
    k_rsqrt<<<blkN, TPB, 0, stream>>>(deg, dinv, N);

    // --- layer 1 ---
    k_cvt_half<<<blkNF, TPB, 0, stream>>>(x, Ah, NF);
    k_cvt_WT<<<blkW, TPB, 0, stream>>>(W1, WT);
    k_zero<<<blkNF, TPB, 0, stream>>>(agg, NF);
    k_gemm_wmma<<<blkGemm, TPB, 0, stream>>>(Ah, WT, h, N);
    k_edge_scatter<<<blkEdge, TPB, 0, stream>>>(src, dst, dinv, h, agg, E);
    k_self_loop<<<blkNF, TPB, 0, stream>>>(dinv, h, agg, NF);
    k_bias_relu<<<blkNF, TPB, 0, stream>>>(agg, b1, h, Ah, NF);  // z -> h (f32) + Ah (f16)

    // --- layer 2 ---
    k_cvt_WT<<<blkW, TPB, 0, stream>>>(W2, WT);
    k_zero<<<blkNF, TPB, 0, stream>>>(agg, NF);
    k_gemm_wmma<<<blkGemm, TPB, 0, stream>>>(Ah, WT, h, N);      // overwrites z, no longer needed
    k_edge_scatter<<<blkEdge, TPB, 0, stream>>>(src, dst, dinv, h, agg, E);
    k_self_loop<<<blkNF, TPB, 0, stream>>>(dinv, h, agg, NF);
    k_bias_relu<<<blkNF, TPB, 0, stream>>>(agg, b2, (float*)d_out, (_Float16*)nullptr, NF);
}
